// UserGraphNet_40157944217636
// MI455X (gfx1250) — matrix operand order repaired and checked by reading, not compiled, not run on hardware
//
#include <hip/hip_runtime.h>
#include <hip/hip_bf16.h>

// ---------------------------------------------------------------------------
// UserGraphNet forward on gfx1250 (MI455X): WMMA bf16 GEMMs + async LDS copy
// + atomic edge ops
// ---------------------------------------------------------------------------

typedef __attribute__((ext_vector_type(16))) __bf16 v16bf;
typedef __attribute__((ext_vector_type(8)))  float  v8f;

constexpr int NN = 714;   // nodes per graph
constexpr int NP = 736;   // nodes padded to 46*16 (also 23*32 for WMMA K)
constexpr int CC = 128;   // channels

static __device__ __forceinline__ float lrelu01(float x) { return x > 0.f ? x : 0.01f * x; }

// order-preserving float <-> uint for atomicMax
static __device__ __forceinline__ unsigned fenc(float f) {
  unsigned u = __float_as_uint(f);
  return (u & 0x80000000u) ? ~u : (u | 0x80000000u);
}
static __device__ __forceinline__ float fdec(unsigned k) {
  return __uint_as_float((k & 0x80000000u) ? (k & 0x7FFFFFFFu) : ~k);
}

static __device__ __forceinline__ void edge_rc(const int* __restrict__ edges,
                                               int b, int ei, int E,
                                               int& r, int& c) {
  if (ei < E) {
    r = edges[((long)b * 2 + 0) * E + ei];
    c = edges[((long)b * 2 + 1) * E + ei];
  } else {
    r = c = ei - E;  // appended self loop
  }
}

// ---------------------------- small utility kernels ------------------------

__global__ void k_fill(float* __restrict__ p, float v, long cnt) {
  long i = (long)blockIdx.x * blockDim.x + threadIdx.x;
  if (i < cnt) p[i] = v;
}

__global__ void k_rsqrt(float* __restrict__ p, long cnt) {
  long i = (long)blockIdx.x * blockDim.x + threadIdx.x;
  if (i < cnt) p[i] = rsqrtf(p[i]);
}

// transpose + convert: src [srcRows, cols] f32 -> dst [cols, dstRows] bf16 (zero pad)
__global__ void k_bf16_T(const float* __restrict__ src, __bf16* __restrict__ dst,
                         int srcRows, int cols, int dstRows) {
  long i = (long)blockIdx.x * blockDim.x + threadIdx.x;
  long total = (long)cols * dstRows;
  if (i >= total) return;
  int cc = (int)(i / dstRows);
  int rr = (int)(i - (long)cc * dstRows);
  dst[i] = (rr < srcRows) ? (__bf16)src[(long)rr * cols + cc] : (__bf16)0.0f;
}

__global__ void k_deg(const int* __restrict__ edges, float* __restrict__ deg,
                      int B, int E) {
  long i = (long)blockIdx.x * blockDim.x + threadIdx.x;
  if (i >= (long)B * E) return;
  int b = (int)(i / E), e = (int)(i - (long)b * E);
  int c = edges[((long)b * 2 + 1) * E + e];
  atomicAdd(&deg[(long)b * NP + c], 1.0f);
}

// --------------------------- WMMA GEMM kernels -----------------------------
// All GEMMs have N = 128: block = 256 threads = 8 waves, wave w owns N-tile w.
// B operand is pre-transposed bf16: BT[n][k] contiguous in k (ldbK).
// A 16xK tile is staged in LDS once per block (async DMA in the generic GEMM).

// Fused embedding-gather + x @ W_in  (K = 800, M tiles of 16 nodes, batch = B)
__global__ void __launch_bounds__(256)
k_embed_gemm(const int* __restrict__ feat,
             const float* __restrict__ poi, const float* __restrict__ cat,
             const float* __restrict__ lat, const float* __restrict__ lng,
             const __bf16* __restrict__ WT /*[128][800]*/,
             float* __restrict__ out /*[B][NP][CC]*/) {
  __shared__ __attribute__((aligned(16))) __bf16 As[16 * 800];
  const int b = blockIdx.y, mt = blockIdx.x;
  // stage gathered+converted embedding rows for this 16-node tile
  for (int i = threadIdx.x; i < 16 * 800; i += 256) {
    int r = i / 800, k = i - r * 800;
    int m = mt * 16 + r;
    int node = (m < NN) ? m : 0;
    const int* f5 = feat + ((long)b * NN + node) * 5;
    float v;
    if (k < 300)      v = poi[(long)f5[0] * 300 + k];
    else if (k < 400) v = cat[(long)f5[1] * 100 + (k - 300)];
    else if (k < 600) v = lat[(long)f5[3] * 200 + (k - 400)];
    else              v = lng[(long)f5[4] * 200 + (k - 600)];
    As[i] = (__bf16)v;
  }
  __syncthreads();

  const int lane = threadIdx.x & 31, half = lane >> 4, lr = lane & 15;
  const int nt = threadIdx.x >> 5;
  const int nn = nt * 16 + lr;
  v8f acc = {};
  for (int k0 = 0; k0 < 800; k0 += 32) {
    v16bf a, bt;
#pragma unroll
    for (int p = 0; p < 8; ++p) {
      int kb = 2 * p + 16 * half;
      a[2 * p]     = As[lr * 800 + k0 + kb];
      a[2 * p + 1] = As[lr * 800 + k0 + kb + 1];
    }
#pragma unroll
    for (int e = 0; e < 16; ++e) bt[e] = WT[(long)nn * 800 + k0 + 16 * half + e];
    acc = __builtin_amdgcn_wmma_f32_16x16x32_bf16(false, a, false, bt,
                                                  (short)0, acc, false, false);
  }
#pragma unroll
  for (int r = 0; r < 8; ++r) {
    int row = mt * 16 + r + 8 * half;
    out[((long)b * NP + row) * CC + nn] = acc[r];
  }
}

// generic batched  D = A(bf16) @ BT(bf16)  with optional bias / leaky-relu / bf16 copy
// A tile staged via GLOBAL_LOAD_ASYNC_TO_LDS_B128 (ASYNCcnt-tracked DMA).
__global__ void __launch_bounds__(256)
k_gemm_n128(const __bf16* __restrict__ A, long sA, int lda,
            const __bf16* __restrict__ BT, long sB, int ldbK,
            float* __restrict__ D, long sD,
            const float* __restrict__ bias, int act,
            __bf16* __restrict__ Dbf, long sDbf, int ksteps) {
  __shared__ __attribute__((aligned(16))) __bf16 As[16 * 800];
  const int b = blockIdx.y;
  A += (long)b * sA;
  BT += (long)b * sB;
  D += (long)b * sD;
  if (Dbf) Dbf += (long)b * sDbf;

  const int K = ksteps * 32;  // <= 800; K multiple of 32 => rows 16B-aligned
  {
    // async copy: 16 rows x K bf16 -> LDS, 16 bytes per lane per issue
    const unsigned long long gbase =
        (unsigned long long)(const char*)(A + (long)(blockIdx.x * 16) * lda);
    const unsigned ldsbase = (unsigned)(unsigned long long)(uintptr_t)&As[0];
    const int chunksPerRow = K / 8;           // 16B chunks per row
    const int total = 16 * chunksPerRow;
    for (int i = threadIdx.x; i < total; i += 256) {
      int r = i / chunksPerRow, cch = i - r * chunksPerRow;
      unsigned long long gaddr = gbase + (unsigned long long)r * (unsigned)(lda * 2)
                                       + (unsigned)(cch * 16);
      unsigned laddr = ldsbase + (unsigned)(i * 16);
      asm volatile("global_load_async_to_lds_b128 %0, %1, off"
                   :: "v"(laddr), "v"(gaddr) : "memory");
    }
    asm volatile("s_wait_asynccnt 0" ::: "memory");
  }
  __syncthreads();

  const int lane = threadIdx.x & 31, half = lane >> 4, lr = lane & 15;
  const int nt = threadIdx.x >> 5;
  const int nn = nt * 16 + lr;
  v8f acc = {};
  for (int ks = 0; ks < ksteps; ++ks) {
    int k0 = ks * 32;
    v16bf a, bt;
#pragma unroll
    for (int p = 0; p < 8; ++p) {
      int kb = 2 * p + 16 * half;
      a[2 * p]     = As[lr * K + k0 + kb];
      a[2 * p + 1] = As[lr * K + k0 + kb + 1];
    }
#pragma unroll
    for (int e = 0; e < 16; ++e) bt[e] = BT[(long)nn * ldbK + k0 + 16 * half + e];
    acc = __builtin_amdgcn_wmma_f32_16x16x32_bf16(false, a, false, bt,
                                                  (short)0, acc, false, false);
  }
#pragma unroll
  for (int r = 0; r < 8; ++r) {
    int row = blockIdx.x * 16 + r + 8 * half;
    float v = acc[r];
    if (bias) v += bias[nn];
    if (act) v = lrelu01(v);
    D[(long)row * CC + nn] = v;
    if (Dbf) Dbf[(long)row * CC + nn] = (__bf16)v;
  }
}

// ------------------------------ edge kernels -------------------------------

// GCN: out[c] += h[r] * dinv[r]*dinv[c]    (one wave32 per edge, 4 ch/lane)
__global__ void __launch_bounds__(256)
k_gcn_agg(const int* __restrict__ edges, const float* __restrict__ hin,
          const float* __restrict__ dinv, float* __restrict__ hout,
          int B, int E, int ET) {
  long wid = ((long)blockIdx.x * blockDim.x + threadIdx.x) >> 5;
  int lane = threadIdx.x & 31;
  if (wid >= (long)B * ET) return;
  int b = (int)(wid / ET), ei = (int)(wid - (long)b * ET);
  int r, c;
  edge_rc(edges, b, ei, E, r, c);
  float nrm = dinv[(long)b * NP + r] * dinv[(long)b * NP + c];
  const float* src = hin + ((long)b * NP + r) * CC;
  float* dst = hout + ((long)b * NP + c) * CC;
  __builtin_prefetch(src, 0, 0);   // global_prefetch: pull gathered row toward WGP
#pragma unroll
  for (int j = 0; j < 4; ++j) {
    int ch = lane + 32 * j;
    atomicAdd(&dst[ch], src[ch] * nrm);
  }
}

// epilogue: v = x + bias; mode0: lrelu(v)  mode1: lrelu(v)+v ; f32 + bf16 out
__global__ void k_post(const float* __restrict__ x, const float* __restrict__ bias,
                       float* __restrict__ oF, __bf16* __restrict__ oBF,
                       int mode, long total) {
  long i = (long)blockIdx.x * blockDim.x + threadIdx.x;
  if (i >= total) return;
  int c = (int)(i % CC);
  float t = x[i] + bias[c];
  float l = lrelu01(t);
  float v = mode ? (l + t) : l;
  oF[i] = v;
  if (oBF) oBF[i] = (__bf16)v;
}

// per-node dot products (wave per node, shuffle reduce); a2/o2 optional
__global__ void __launch_bounds__(256)
k_dots(const float* __restrict__ t, const float* __restrict__ a1,
       const float* __restrict__ a2, float* __restrict__ o1,
       float* __restrict__ o2, int B) {
  long wid = ((long)blockIdx.x * blockDim.x + threadIdx.x) >> 5;
  int lane = threadIdx.x & 31;
  if (wid >= (long)B * NN) return;
  int b = (int)(wid / NN), node = (int)(wid - (long)b * NN);
  const float* row = t + ((long)b * NP + node) * CC;
  float s1 = 0.f, s2 = 0.f;
#pragma unroll
  for (int j = 0; j < 4; ++j) {
    int ch = lane + 32 * j;
    float v = row[ch];
    s1 += v * a1[ch];
    if (a2) s2 += v * a2[ch];
  }
#pragma unroll
  for (int off = 16; off; off >>= 1) {
    s1 += __shfl_xor(s1, off, 32);
    if (a2) s2 += __shfl_xor(s2, off, 32);
  }
  if (lane == 0) {
    o1[(long)b * NP + node] = s1;
    if (a2) o2[(long)b * NP + node] = s2;
  }
}

// GAT softmax pass 1: e = leaky_relu(asrc[r]+adst[c], 0.2); segment max over c
__global__ void k_edge_score(const int* __restrict__ edges,
                             const float* __restrict__ asrc,
                             const float* __restrict__ adst,
                             float* __restrict__ eE, unsigned* __restrict__ emaxU,
                             int B, int E, int ET) {
  long i = (long)blockIdx.x * blockDim.x + threadIdx.x;
  if (i >= (long)B * ET) return;
  int b = (int)(i / ET), ei = (int)(i - (long)b * ET);
  int r, c;
  edge_rc(edges, b, ei, E, r, c);
  float e = asrc[(long)b * NP + r] + adst[(long)b * NP + c];
  e = e > 0.f ? e : 0.2f * e;
  eE[i] = e;
  atomicMax(&emaxU[(long)b * NP + c], fenc(e));
}

// GAT softmax pass 2: ee = exp(e - max[c]); segment sum over c
__global__ void k_edge_expsum(const int* __restrict__ edges,
                              float* __restrict__ eE,
                              const unsigned* __restrict__ emaxU,
                              float* __restrict__ denom, int B, int E, int ET) {
  long i = (long)blockIdx.x * blockDim.x + threadIdx.x;
  if (i >= (long)B * ET) return;
  int b = (int)(i / ET), ei = (int)(i - (long)b * ET);
  int r, c;
  edge_rc(edges, b, ei, E, r, c);
  float x = __expf(eE[i] - fdec(emaxU[(long)b * NP + c]));
  eE[i] = x;
  atomicAdd(&denom[(long)b * NP + c], x);
}

// GAT aggregate: out[c] += t[r] * alpha   (wave per edge)
__global__ void __launch_bounds__(256)
k_gat_agg(const int* __restrict__ edges, const float* __restrict__ eE,
          const float* __restrict__ denom, const float* __restrict__ t,
          float* __restrict__ out, int B, int E, int ET) {
  long wid = ((long)blockIdx.x * blockDim.x + threadIdx.x) >> 5;
  int lane = threadIdx.x & 31;
  if (wid >= (long)B * ET) return;
  int b = (int)(wid / ET), ei = (int)(wid - (long)b * ET);
  int r, c;
  edge_rc(edges, b, ei, E, r, c);
  float alpha = eE[wid] / (denom[(long)b * NP + c] + 1e-16f);
  const float* src = t + ((long)b * NP + r) * CC;
  float* dst = out + ((long)b * NP + c) * CC;
  __builtin_prefetch(src, 0, 0);
#pragma unroll
  for (int j = 0; j < 4; ++j) {
    int ch = lane + 32 * j;
    atomicAdd(&dst[ch], src[ch] * alpha);
  }
}

// scalar GCN aggregate for the C->1 output conv
__global__ void k_edge_sagg(const int* __restrict__ edges,
                            const float* __restrict__ sN,
                            const float* __restrict__ dinv,
                            float* __restrict__ g, int B, int E, int ET) {
  long i = (long)blockIdx.x * blockDim.x + threadIdx.x;
  if (i >= (long)B * ET) return;
  int b = (int)(i / ET), ei = (int)(i - (long)b * ET);
  int r, c;
  edge_rc(edges, b, ei, E, r, c);
  float v = sN[(long)b * NP + r] * dinv[(long)b * NP + r] * dinv[(long)b * NP + c];
  atomicAdd(&g[(long)b * NP + c], v);
}

// g = lrelu(gagg + b_out); zero the K-padding columns so FC1 GEMM is exact
__global__ void k_gpost(const float* __restrict__ gagg, const float* __restrict__ bout,
                        float* __restrict__ gF, __bf16* __restrict__ gBF, int B) {
  long i = (long)blockIdx.x * blockDim.x + threadIdx.x;
  if (i >= (long)B * NP) return;
  int node = (int)(i % NP);
  float v = 0.f;
  if (node < NN) v = lrelu01(gagg[i] + bout[0]);
  gF[i] = v;
  gBF[i] = (__bf16)v;
}

// ------------------------------- launcher ----------------------------------

extern "C" void kernel_launch(void* const* d_in, const int* in_sizes, int n_in,
                              void* d_out, int out_size, void* d_ws, size_t ws_size,
                              hipStream_t stream) {
  const int*   feature  = (const int*)d_in[0];
  const int*   edges    = (const int*)d_in[1];
  // d_in[2] (edge weight) is unused by the reference math
  const float* poi      = (const float*)d_in[3];
  const float* cat      = (const float*)d_in[4];
  const float* lat      = (const float*)d_in[5];
  const float* lng      = (const float*)d_in[6];
  const float* W_in     = (const float*)d_in[7];
  const float* b_in     = (const float*)d_in[8];
  const float* gat_W    = (const float*)d_in[9];
  const float* gat_asrc = (const float*)d_in[10];
  const float* gat_adst = (const float*)d_in[11];
  const float* gat_b    = (const float*)d_in[12];
  const float* W_out    = (const float*)d_in[13];
  const float* b_out    = (const float*)d_in[14];
  const float* W_fc1    = (const float*)d_in[15];
  const float* b_fc1    = (const float*)d_in[16];
  const float* W_fc2    = (const float*)d_in[17];
  const float* b_fc2    = (const float*)d_in[18];

  const int B  = in_sizes[0] / (NN * 5);
  const int E  = in_sizes[1] / (2 * B);
  const int ET = E + NN;

  // ---- workspace carve-up (256B aligned) ----
  char* wp = (char*)d_ws;
  auto alloc = [&](size_t bytes) -> void* {
    void* p = (void*)wp;
    wp += (bytes + 255) & ~(size_t)255;
    return p;
  };
  __bf16* winBT  = (__bf16*)alloc((size_t)CC * 800 * 2);      // [128][800]
  __bf16* gatWBT = (__bf16*)alloc((size_t)3 * CC * CC * 2);   // 3x [128][128]
  __bf16* fc1BT  = (__bf16*)alloc((size_t)CC * NP * 2);       // [128][736]
  __bf16* fc2BT  = (__bf16*)alloc((size_t)CC * CC * 2);       // [128][128]
  float*  hF     = (float*)alloc((size_t)B * NP * CC * 4);
  __bf16* hBF    = (__bf16*)alloc((size_t)B * NP * CC * 2);
  float*  tF     = (float*)alloc((size_t)B * NP * CC * 4);
  float*  aggF   = (float*)alloc((size_t)B * NP * CC * 4);
  float*  dinv   = (float*)alloc((size_t)B * NP * 4);
  float*  asrcN  = (float*)alloc((size_t)B * NP * 4);
  float*  adstN  = (float*)alloc((size_t)B * NP * 4);
  unsigned* emaxU= (unsigned*)alloc((size_t)B * NP * 4);
  float*  denomF = (float*)alloc((size_t)B * NP * 4);
  float*  eEdge  = (float*)alloc((size_t)B * ET * 4);
  float*  sN     = (float*)alloc((size_t)B * NP * 4);
  float*  gaggF  = (float*)alloc((size_t)B * NP * 4);
  float*  gF     = (float*)alloc((size_t)B * NP * 4);
  __bf16* gBF    = (__bf16*)alloc((size_t)B * NP * 2);
  float*  t1F    = (float*)alloc((size_t)B * CC * 4);
  __bf16* t1BF   = (__bf16*)alloc((size_t)B * CC * 2);

  const dim3 blk(256);
  auto g1 = [](long t) { return dim3((unsigned)((t + 255) / 256)); };

  // ---- weight convert+transpose to bf16 ----
  k_bf16_T<<<g1((long)CC * 800), blk, 0, stream>>>(W_in, winBT, 800, CC, 800);
  for (int u = 0; u < 3; ++u)
    k_bf16_T<<<g1((long)CC * CC), blk, 0, stream>>>(gat_W + (long)u * CC * CC,
                                                    gatWBT + (long)u * CC * CC, CC, CC, CC);
  k_bf16_T<<<g1((long)CC * NP), blk, 0, stream>>>(W_fc1, fc1BT, NN, CC, NP);
  k_bf16_T<<<g1((long)CC * CC), blk, 0, stream>>>(W_fc2, fc2BT, CC, CC, CC);

  // ---- symmetric-norm degrees (self loop => init 1.0) ----
  k_fill<<<g1((long)B * NP), blk, 0, stream>>>(dinv, 1.0f, (long)B * NP);
  k_deg<<<g1((long)B * E), blk, 0, stream>>>(edges, dinv, B, E);
  k_rsqrt<<<g1((long)B * NP), blk, 0, stream>>>(dinv, (long)B * NP);

  // ---- GCN input layer: fused embed-gather + WMMA GEMM, then aggregate ----
  k_embed_gemm<<<dim3(NP / 16, B), blk, 0, stream>>>(feature, poi, cat, lat, lng, winBT, tF);
  k_fill<<<g1((long)B * NP * CC), blk, 0, stream>>>(aggF, 0.f, (long)B * NP * CC);
  k_gcn_agg<<<g1((long)B * ET * 32), blk, 0, stream>>>(edges, tF, dinv, aggF, B, E, ET);
  k_post<<<g1((long)B * NP * CC), blk, 0, stream>>>(aggF, b_in, hF, hBF, 0, (long)B * NP * CC);

  // ---- 3 AGCN units, each applies its GATConv twice ----
  for (int step = 0; step < 6; ++step) {
    int u = step >> 1, second = step & 1;
    k_gemm_n128<<<dim3(NP / 16, B), blk, 0, stream>>>(
        hBF, (long)NP * CC, CC, gatWBT + (long)u * CC * CC, 0L, CC,
        tF, (long)NP * CC, nullptr, 0, nullptr, 0L, CC / 32);
    k_dots<<<g1((long)B * NN * 32), blk, 0, stream>>>(tF, gat_asrc + u * CC,
                                                      gat_adst + u * CC, asrcN, adstN, B);
    k_fill<<<g1((long)B * NP), blk, 0, stream>>>((float*)emaxU, 0.f, (long)B * NP);
    k_fill<<<g1((long)B * NP), blk, 0, stream>>>(denomF, 0.f, (long)B * NP);
    k_edge_score<<<g1((long)B * ET), blk, 0, stream>>>(edges, asrcN, adstN, eEdge, emaxU, B, E, ET);
    k_edge_expsum<<<g1((long)B * ET), blk, 0, stream>>>(edges, eEdge, emaxU, denomF, B, E, ET);
    k_fill<<<g1((long)B * NP * CC), blk, 0, stream>>>(aggF, 0.f, (long)B * NP * CC);
    k_gat_agg<<<g1((long)B * ET * 32), blk, 0, stream>>>(edges, eEdge, denomF, tF, aggF, B, E, ET);
    // first call of pair: h = lrelu(t)+t ; second: h = lrelu(t)
    k_post<<<g1((long)B * NP * CC), blk, 0, stream>>>(aggF, gat_b + u * CC, hF, hBF,
                                                      second ? 0 : 1, (long)B * NP * CC);
  }

  // ---- output GCN (128 -> 1): per-node dot, scalar aggregate, lrelu ----
  k_dots<<<g1((long)B * NN * 32), blk, 0, stream>>>(hF, W_out, nullptr, sN, nullptr, B);
  k_fill<<<g1((long)B * NP), blk, 0, stream>>>(gaggF, 0.f, (long)B * NP);
  k_edge_sagg<<<g1((long)B * ET), blk, 0, stream>>>(edges, sN, dinv, gaggF, B, E, ET);
  k_gpost<<<g1((long)B * NP), blk, 0, stream>>>(gaggF, b_out, gF, gBF, B);

  // ---- FC head: [B x 736] @ [736 x 128] (lrelu) then [B x 128] @ [128 x 128] ----
  k_gemm_n128<<<dim3((B + 15) / 16, 1), blk, 0, stream>>>(
      gBF, 0L, NP, fc1BT, 0L, NP, t1F, 0L, b_fc1, 1, t1BF, 0L, NP / 32);
  k_gemm_n128<<<dim3((B + 15) / 16, 1), blk, 0, stream>>>(
      t1BF, 0L, CC, fc2BT, 0L, CC, (float*)d_out, 0L, b_fc2, 0, nullptr, 0L, CC / 32);
}
